// ModulatedLinear2_39444979646806
// MI455X (gfx1250) — matrix operand order repaired and checked by reading, not compiled, not run on hardware
//
#include <hip/hip_runtime.h>
#include <hip/hip_bf16.h>

// ---------------------------------------------------------------------------
// ModulatedLinear2: per-batch demodulated weight (384x384) x X (384x4096)
// bf16 WMMA path (v_wmma_f32_16x16x32_bf16), f32 accumulate.
// Double-buffered LDS pipeline; async global->LDS (ASYNCcnt) for the bf16 A
// tile; RNE bf16 packing; non-temporal output stores.
// ---------------------------------------------------------------------------

typedef __attribute__((ext_vector_type(16))) __bf16 v16bf;
typedef __attribute__((ext_vector_type(8)))  __bf16 v8bf;
typedef __attribute__((ext_vector_type(2)))  __bf16 v2bf;
typedef __attribute__((ext_vector_type(8)))  float  v8f;

static constexpr int NB   = 16;     // batches
static constexpr int LL   = 4096;   // sequence length (N of GEMM)
static constexpr int INF  = 384;    // K of GEMM
static constexpr int OUTF = 384;    // M of GEMM
static constexpr float SCALE_C = 0.05103103630798288f; // 1/sqrt(384)
static constexpr float EPS_C   = 1e-8f;

#if __has_builtin(__builtin_amdgcn_global_load_async_to_lds_b128)
#define ASYNC_LDS 1
typedef int v4i_b __attribute__((vector_size(16)));
typedef __attribute__((address_space(3))) v4i_b* as3_v4i_p;
#endif

__device__ __forceinline__ void async_fence() {
#if defined(ASYNC_LDS)
#if __has_builtin(__builtin_amdgcn_s_wait_asynccnt)
  __builtin_amdgcn_s_wait_asynccnt(0);
#else
  asm volatile("s_wait_asynccnt 0" ::: "memory");
#endif
#endif
}

__device__ __forceinline__ unsigned short f32_to_bf16_rne(float f) {
  unsigned u = __float_as_uint(f);
  unsigned r = u + 0x7FFFu + ((u >> 16) & 1u);
  return (unsigned short)(r >> 16);
}

__device__ __forceinline__ unsigned pack_bf16x2(float lo, float hi) {
#if __has_builtin(__builtin_amdgcn_cvt_pk_bf16_f32)
  auto p = __builtin_amdgcn_cvt_pk_bf16_f32(lo, hi);   // packed RNE cvt
  return __builtin_bit_cast(unsigned, p);
#else
  // Native fptrunc: backend picks the best gfx1250 lowering (RNE either way).
  v2bf p;
  p[0] = (__bf16)lo;
  p[1] = (__bf16)hi;
  return __builtin_bit_cast(unsigned, p);
#endif
}

// ---------------------------------------------------------------------------
// Kernel 1: W_bf16[b][o][i] = bf16( dconf(b,o) * SCALE * weight[o][i] * y[b][i] )
// One wave32 per (b,o) row.
// ---------------------------------------------------------------------------
__global__ __launch_bounds__(256)
void modw_kernel(const float* __restrict__ weight,
                 const float* __restrict__ y,
                 unsigned short* __restrict__ Wbf) {
  const int row  = (blockIdx.x * blockDim.x + threadIdx.x) >> 5; // wave id
  const int lane = threadIdx.x & 31;
  const int b = row / OUTF;
  const int o = row % OUTF;

  const float* wrow = weight + (size_t)o * INF;
  const float* yrow = y + (size_t)b * INF;

  float vals[INF / 32];
  float ss = 0.f;
#pragma unroll
  for (int j = 0; j < INF / 32; ++j) {
    const int i = lane + 32 * j;
    const float v = SCALE_C * wrow[i] * yrow[i];
    vals[j] = v;
    ss += v * v;
  }
#pragma unroll
  for (int m = 16; m >= 1; m >>= 1) ss += __shfl_xor(ss, m, 32);
  const float d = rsqrtf(ss + EPS_C);

  unsigned short* orow = Wbf + (size_t)(b * OUTF + o) * INF;
#pragma unroll
  for (int j = 0; j < INF / 32; ++j)
    orow[lane + 32 * j] = f32_to_bf16_rne(vals[j] * d);
}

// ---------------------------------------------------------------------------
// Kernel 2: Out_b[384x4096] = W_b[384x384] * X_b[384x4096]
// Block tile: 128(M) x 128(N), 8 waves (2x4), wave tile 64x32, K-step 32.
// Double-buffered LDS (40 KB total), 1 barrier per k-step, no loop branches.
// ---------------------------------------------------------------------------
static constexpr int LDA_S = 40;  // halves per A row -> 16B aligned, conflict-free
static constexpr int LDB_S = 40;  // halves per B col

struct Bregs { float4 r0, r1, r2, r3; };

__global__ __launch_bounds__(256)
void mgemm_kernel(const float* __restrict__ X,          // Efou, [B][384][4096] raw view
                  const unsigned short* __restrict__ Wbf,
                  float* __restrict__ Out) {
  __shared__ unsigned short lA[2][128 * LDA_S];  // 2 x 10240 B
  __shared__ unsigned short lB[2][128 * LDB_S];  // 2 x 10240 B

  const int tid   = threadIdx.x;
  const int lane  = tid & 31;
  const int wave  = tid >> 5;
  const int waveM = wave >> 2;   // 0..1
  const int waveN = wave & 3;    // 0..3

  const int b  = blockIdx.z;
  const int m0 = blockIdx.y * 128;
  const int n0 = blockIdx.x * 128;

  const float* Xb = X + (size_t)b * INF * LL;
  const unsigned short* Wb = Wbf + (size_t)b * OUTF * INF;
  float* Cb = Out + (size_t)b * OUTF * LL;

  v8f acc[4][2];
#pragma unroll
  for (int i = 0; i < 4; ++i)
#pragma unroll
    for (int j = 0; j < 2; ++j)
#pragma unroll
      for (int v = 0; v < 8; ++v) acc[i][j][v] = 0.f;

  // staging decomposition
  const int a_row  = tid >> 1;          // 0..127
  const int a_half = (tid & 1) * 16;    // 0 or 16 halves within the 32-wide K slab
  const int b_c4   = (tid & 31) * 4;    // column group (0..124)
  const int b_g    = tid >> 5;          // 0..7 -> k-pairs {2g,2g+1} and {16+2g,17+2g}

  const int lm = lane & 15;             // M-row / N-col within a 16-tile
  const int lc = lane >> 4;             // lane half selector

  // ---- staging helpers -----------------------------------------------------
  auto stageA = [&](int buf, int k0) {
    const unsigned short* src = Wb + (size_t)(m0 + a_row) * INF + k0 + a_half;
    unsigned short* dst = &lA[buf][a_row * LDA_S + a_half];
#if defined(ASYNC_LDS)
    __builtin_amdgcn_global_load_async_to_lds_b128(
        (v4i_b*)src, (as3_v4i_p)dst, 0, 0);
    __builtin_amdgcn_global_load_async_to_lds_b128(
        (v4i_b*)(src + 8), (as3_v4i_p)(dst + 8), 0, 0);
#else
    *(uint4*)(dst)     = *(const uint4*)(src);
    *(uint4*)(dst + 8) = *(const uint4*)(src + 8);
#endif
  };

  auto loadB = [&](int k0) -> Bregs {
    const float* xs = Xb + (size_t)(k0 + 2 * b_g) * LL + n0 + b_c4;
    Bregs r;
    r.r0 = *(const float4*)(xs);
    r.r1 = *(const float4*)(xs + LL);
    r.r2 = *(const float4*)(xs + 16 * LL);
    r.r3 = *(const float4*)(xs + 17 * LL);
    return r;
  };

  auto storeB = [&](int buf, const Bregs& r) {
#pragma unroll
    for (int c = 0; c < 4; ++c) {
      const float a0 = (&r.r0.x)[c], a1 = (&r.r1.x)[c];
      const float a2 = (&r.r2.x)[c], a3 = (&r.r3.x)[c];
      const int col = b_c4 + c;
      *(unsigned*)&lB[buf][col * LDB_S + 2 * b_g]      = pack_bf16x2(a0, a1);
      *(unsigned*)&lB[buf][col * LDB_S + 16 + 2 * b_g] = pack_bf16x2(a2, a3);
    }
  };

  auto compute = [&](int cbuf) {
    v16bf afr[4];
#pragma unroll
    for (int i = 0; i < 4; ++i) {
      const int row = waveM * 64 + i * 16 + lm;
      const v8bf x0 = *(const v8bf*)&lA[cbuf][row * LDA_S + lc * 8];       // K = c*8..+7
      const v8bf x1 = *(const v8bf*)&lA[cbuf][row * LDA_S + 16 + lc * 8];  // K = 16+c*8..
      afr[i] = __builtin_shufflevector(x0, x1, 0, 1, 2, 3, 4, 5, 6, 7,
                                       8, 9, 10, 11, 12, 13, 14, 15);
    }
    v16bf bfr[2];
#pragma unroll
    for (int j = 0; j < 2; ++j) {
      const int col = waveN * 32 + j * 16 + lm;
      const v8bf y0 = *(const v8bf*)&lB[cbuf][col * LDB_S + lc * 8];
      const v8bf y1 = *(const v8bf*)&lB[cbuf][col * LDB_S + 16 + lc * 8];
      bfr[j] = __builtin_shufflevector(y0, y1, 0, 1, 2, 3, 4, 5, 6, 7,
                                       8, 9, 10, 11, 12, 13, 14, 15);
    }
#pragma unroll
    for (int i = 0; i < 4; ++i)
#pragma unroll
      for (int j = 0; j < 2; ++j)
        acc[i][j] = __builtin_amdgcn_wmma_f32_16x16x32_bf16(
            /*neg_a=*/false, afr[i], /*neg_b=*/false, bfr[j],
            /*c_mod=*/(short)0, acc[i][j], /*reuse_a=*/false, /*reuse_b=*/false);
  };

  // ---- prologue: fill buffer 0 --------------------------------------------
  {
    stageA(0, 0);
    const Bregs r = loadB(0);
    storeB(0, r);
  }
  async_fence();
  __syncthreads();

  // ---- steady state: 11 iterations, stage-next + compute-current ----------
  int buf = 0;
  for (int k0 = 0; k0 < INF - 32; k0 += 32, buf ^= 1) {
    stageA(buf ^ 1, k0 + 32);        // async copies into the other buffer
    const Bregs rn = loadB(k0 + 32); // f32 loads in flight during the WMMAs
    compute(buf);
    storeB(buf ^ 1, rn);
    async_fence();                   // my async A writes have landed in LDS
    __syncthreads();                 // everyone's have
  }

  // ---- final tile ----------------------------------------------------------
  compute(buf);

  // ---- epilogue: D layout -> lane = N col, VGPR v = M row (v + 8*(lane>>4))
  // Output is write-once: non-temporal stores keep X tiles resident in L2.
#pragma unroll
  for (int i = 0; i < 4; ++i) {
#pragma unroll
    for (int j = 0; j < 2; ++j) {
      const int mg = m0 + waveM * 64 + i * 16 + lc * 8;
      const int ng = n0 + waveN * 32 + j * 16 + lm;
      float* cp = Cb + (size_t)mg * LL + ng;
#pragma unroll
      for (int v = 0; v < 8; ++v)
        __builtin_nontemporal_store(acc[i][j][v], cp + (size_t)v * LL);
    }
  }
}

// ---------------------------------------------------------------------------
extern "C" void kernel_launch(void* const* d_in, const int* in_sizes, int n_in,
                              void* d_out, int out_size, void* d_ws, size_t ws_size,
                              hipStream_t stream) {
  const float* Efou   = (const float*)d_in[0];  // [16][4096][384] f32
  const float* y      = (const float*)d_in[1];  // [16][384] f32
  const float* weight = (const float*)d_in[2];  // [1][384][384] f32
  float* out = (float*)d_out;                   // raw view of [16][384][4096] f32

  unsigned short* Wbf = (unsigned short*)d_ws;  // 16*384*384 bf16 = 4.7 MB

  modw_kernel<<<dim3((NB * OUTF * 32) / 256), dim3(256), 0, stream>>>(weight, y, Wbf);
  mgemm_kernel<<<dim3(LL / 128, OUTF / 128, NB), dim3(256), 0, stream>>>(Efou, Wbf, out);
}
